// SelfAttention2d_87050397155874
// MI455X (gfx1250) — matrix operand order repaired and checked by reading, not compiled, not run on hardware
//
#include <hip/hip_runtime.h>

typedef _Float16 half_t;
typedef __attribute__((ext_vector_type(16))) _Float16 v16h;
typedef __attribute__((ext_vector_type(8)))  _Float16 v8h;
typedef __attribute__((ext_vector_type(8)))  float    v8f;

#define Bn 4
#define Cc 512
#define Nn 4096
#define Gg 32
#define CPG 16
#define EPS 1e-5f
#define CNs ((size_t)Cc * Nn)

// LDS tile geometry: BK=32 halves/row (16 DWORDs) + 4 DWORDs pad -> 40-half rows
#define LROW 40

#if defined(__gfx1250__) && __has_builtin(__builtin_amdgcn_tensor_load_to_lds) && \
    __has_builtin(__builtin_amdgcn_s_wait_tensorcnt)
#define HAVE_TDM 1
#else
#define HAVE_TDM 0
#endif

// ---------------------------------------------------------------------------
// block-wide reduction (wave32 shuffles + LDS across 8 waves)
// ---------------------------------------------------------------------------
__device__ __forceinline__ float block_reduce(float v, bool is_max) {
    #pragma unroll
    for (int off = 16; off > 0; off >>= 1) {
        float o = __shfl_xor(v, off, 32);
        v = is_max ? fmaxf(v, o) : (v + o);
    }
    __shared__ float sh[8];
    int wid = threadIdx.x >> 5, lid = threadIdx.x & 31;
    __syncthreads();
    if (lid == 0) sh[wid] = v;
    __syncthreads();
    if (wid == 0) {
        float t = sh[lid & 7];
        #pragma unroll
        for (int off = 4; off > 0; off >>= 1) {
            float o = __shfl_xor(t, off, 32);
            t = is_max ? fmaxf(t, o) : (t + o);
        }
        if (lid == 0) sh[0] = t;
    }
    __syncthreads();
    return sh[0];
}

// ---------------------------------------------------------------------------
// GroupNorm stats
// ---------------------------------------------------------------------------
__global__ __launch_bounds__(256) void gn_stats(const float* __restrict__ x,
                                                float* __restrict__ mean,
                                                float* __restrict__ rstd) {
    int bg = blockIdx.x;
    const float* xs = x + (size_t)bg * CPG * Nn;
    float s = 0.f, ss = 0.f;
    for (int i = threadIdx.x; i < CPG * Nn; i += blockDim.x) {
        float v = xs[i];
        s += v; ss += v * v;
    }
    s  = block_reduce(s,  false);
    ss = block_reduce(ss, false);
    if (threadIdx.x == 0) {
        const float inv_n = 1.0f / (float)(CPG * Nn);
        float m = s * inv_n;
        mean[bg] = m;
        rstd[bg] = rsqrtf(ss * inv_n - m * m + EPS);
    }
}

// ---------------------------------------------------------------------------
// GroupNorm apply -> Zt[b][n][c] f16 (transposed so GEMM B-operand is [N][K])
// ---------------------------------------------------------------------------
__global__ __launch_bounds__(256) void gn_apply(const float* __restrict__ x,
                                                const float* __restrict__ mean,
                                                const float* __restrict__ rstd,
                                                const float* __restrict__ gamma,
                                                const float* __restrict__ beta,
                                                half_t* __restrict__ Zt) {
    size_t idx = (size_t)blockIdx.x * blockDim.x + threadIdx.x;
    if (idx >= (size_t)Bn * CNs) return;
    int n = (int)(idx % Nn);
    size_t cb = idx / Nn;
    int c = (int)(cb % Cc);
    int b = (int)(cb / Cc);
    int bg = b * Gg + c / CPG;
    float z = (x[idx] - mean[bg]) * rstd[bg] * gamma[c] + beta[c];
    Zt[(size_t)b * CNs + (size_t)n * Cc + c] = (half_t)z;
}

__global__ __launch_bounds__(256) void f32_to_f16(const float* __restrict__ s,
                                                  half_t* __restrict__ d, int n) {
    int i = blockIdx.x * blockDim.x + threadIdx.x;
    if (i < n) d[i] = (half_t)s[i];
}

// ---------------------------------------------------------------------------
// TDM: 2-D tile load, tile_dim0 = 32 f16 (16 DWORDs), pad 4 DWORDs per row
// -> 40-half LDS rows. D# per CDNA5 ISA 8.3/8.4 (6-arg clang-23 builtin).
// Group-1 descriptor is loop-invariant and built once per kernel.
// ---------------------------------------------------------------------------
#if HAVE_TDM
typedef __attribute__((ext_vector_type(4))) unsigned int u32x4_t;
typedef __attribute__((ext_vector_type(8))) int i32x8_t;
typedef __attribute__((ext_vector_type(4))) int i32x4_t;

__device__ __forceinline__ i32x8_t tdm_make_g1(unsigned strideElems,
                                               unsigned tensorRows) {
    i32x8_t g1;
    g1[0] = (int)((1u << 16) |            // data_size = 2B
                  (1u << 20) |            // pad_enable
                  (3u << 22) |            // pad_interval: 16 DWORDs
                  (3u << 25));            // pad_amount: 4 DWORDs
    g1[1] = (int)((strideElems & 0xFFFFu) << 16);               // tensor_dim0 lo16
    g1[2] = (int)((strideElems >> 16) | ((tensorRows & 0xFFFFu) << 16));
    g1[3] = (int)((tensorRows >> 16) | (32u << 16));            // tile_dim0 = 32
    g1[4] = (int)128;                                           // tile_dim1 = 128
    g1[5] = (int)strideElems;                                   // dim0_stride lo32
    g1[6] = 0;
    g1[7] = 0;
    return g1;
}

__device__ __forceinline__ void tdm_load_2d(unsigned lds_off, const half_t* g,
                                            i32x8_t g1) {
    unsigned long long ga = (unsigned long long)(uintptr_t)g;
    u32x4_t g0;
    g0.x = 1u;                                                  // count=1 (valid)
    g0.y = lds_off;                                             // lds_addr
    g0.z = (unsigned)ga;                                        // global_addr[31:0]
    g0.w = (unsigned)((ga >> 32) & 0x01FFFFFFull) | (2u << 30); // [56:32] | type=2
    i32x4_t z4 = {0, 0, 0, 0};
    i32x8_t z8 = {0, 0, 0, 0, 0, 0, 0, 0};
    __builtin_amdgcn_tensor_load_to_lds(g0, g1, z4, z4, z8, 0);
}
#endif

// ---------------------------------------------------------------------------
// WMMA tile compute, wave tile 32x64: load 2 A frags + 4 B frags, then issue
// 8 WMMAs back-to-back (12 ds_load_b128 per 8 WMMAs).
// ---------------------------------------------------------------------------
__device__ __forceinline__ void mma_tile(const half_t* bufA, const half_t* bufB,
                                         int wm, int wn, int lrow, int lhi,
                                         v8f acc[2][4]) {
    v16h af[2];
    #pragma unroll
    for (int mt = 0; mt < 2; mt++) {
        const half_t* pa = bufA + (wm * 32 + mt * 16 + lrow) * LROW;
        int ka = lhi * 8;
        v8h alo = *(const v8h*)(pa + ka);
        v8h ahi = *(const v8h*)(pa + ka + 16);
        #pragma unroll
        for (int q = 0; q < 8; q++) { af[mt][q] = alo[q]; af[mt][8 + q] = ahi[q]; }
    }
    v16h bf[4];
    #pragma unroll
    for (int nt = 0; nt < 4; nt++) {
        const half_t* pb = bufB + (wn * 64 + nt * 16 + lrow) * LROW;
        int kb = lhi * 8;
        v8h b0 = *(const v8h*)(pb + kb);
        v8h b1 = *(const v8h*)(pb + kb + 16);
        #pragma unroll
        for (int q = 0; q < 8; q++) { bf[nt][q] = b0[q]; bf[nt][8 + q] = b1[q]; }
    }
    #pragma unroll
    for (int mt = 0; mt < 2; mt++)
        #pragma unroll
        for (int nt = 0; nt < 4; nt++)
            acc[mt][nt] = __builtin_amdgcn_wmma_f32_16x16x32_f16(
                false, af[mt], false, bf[nt], (short)0, acc[mt][nt], false, false);
}

// ---------------------------------------------------------------------------
// Uniform WMMA GEMM: Out[m][n] = scale * sum_k A[m][k] * B[n][k]
//   A stored [M][K] lda=K, B stored [N][K] ldb=K.
//   OUT16: f16 out; EPILOGUE: f32 out + bias[m] + residual; TRANSOUT: Out[n][m].
// Block tile 128x128, K-tile 32, 8 waves (4M x 2N), wave tile 32x64,
// TDM double-buffered staging (40 KB LDS).
// ---------------------------------------------------------------------------
template <bool OUT16, bool EPILOGUE, bool TRANSOUT>
__global__ __launch_bounds__(256)
void gemm_wmma(const half_t* __restrict__ A, const half_t* __restrict__ Bm,
               void* __restrict__ Out, int M, int Nc, int Kd,
               long long aBatch, long long bBatch, long long oBatch,
               float outScale, const float* __restrict__ bias,
               const float* __restrict__ resid, long long rBatch) {
    __shared__ half_t sA[2][128 * LROW];
    __shared__ half_t sB[2][128 * LROW];

    const int bz = blockIdx.z;
    const half_t* Ab = A  + (size_t)bz * aBatch;
    const half_t* Bb = Bm + (size_t)bz * bBatch;
    const int m0 = blockIdx.y * 128;
    const int n0 = blockIdx.x * 128;
    const int tid = threadIdx.x;
    const int wid = tid >> 5, lane = tid & 31;
    const int wm = wid & 3, wn = wid >> 2;
    const int lrow = lane & 15, lhi = lane >> 4;

    const half_t* gA = Ab + (size_t)m0 * Kd;
    const half_t* gB = Bb + (size_t)n0 * Kd;

    v8f acc[2][4] = {};

#if HAVE_TDM
    // ---- async TDM double-buffered pipeline ----
    const i32x8_t g1A = tdm_make_g1((unsigned)Kd, (unsigned)M);
    const i32x8_t g1B = tdm_make_g1((unsigned)Kd, (unsigned)Nc);
    if (wid == 0) {
        tdm_load_2d((unsigned)(uintptr_t)&sA[0][0], gA, g1A);
        tdm_load_2d((unsigned)(uintptr_t)&sB[0][0], gB, g1B);
    }
    int p = 0;
    for (int k0 = 0; k0 < Kd; k0 += 32) {
        if (wid == 0) {
            if (k0 + 32 < Kd) {
                tdm_load_2d((unsigned)(uintptr_t)&sA[p ^ 1][0], gA + k0 + 32, g1A);
                tdm_load_2d((unsigned)(uintptr_t)&sB[p ^ 1][0], gB + k0 + 32, g1B);
                __builtin_amdgcn_s_wait_tensorcnt(2);   // current tile landed
            } else {
                __builtin_amdgcn_s_wait_tensorcnt(0);
            }
        }
        __syncthreads();
        mma_tile(&sA[p][0], &sB[p][0], wm, wn, lrow, lhi, acc);
        __syncthreads();
        p ^= 1;
    }
#else
    // ---- manual register-pipelined staging (all b128) ----
    v8h ar[2], br[2];
    #pragma unroll
    for (int c = 0; c < 2; c++) {
        int id = c * 256 + tid, row = id >> 2, koff = (id & 3) * 8;
        ar[c] = *(const v8h*)(gA + (size_t)row * Kd + koff);
        br[c] = *(const v8h*)(gB + (size_t)row * Kd + koff);
    }
    for (int k0 = 0; k0 < Kd; k0 += 32) {
        #pragma unroll
        for (int c = 0; c < 2; c++) {
            int id = c * 256 + tid, row = id >> 2, koff = (id & 3) * 8;
            *(v8h*)&sA[0][row * LROW + koff] = ar[c];
            *(v8h*)&sB[0][row * LROW + koff] = br[c];
        }
        __syncthreads();
        if (k0 + 32 < Kd) {          // next tile loads overlap compute
            #pragma unroll
            for (int c = 0; c < 2; c++) {
                int id = c * 256 + tid, row = id >> 2, koff = (id & 3) * 8;
                ar[c] = *(const v8h*)(gA + (size_t)row * Kd + k0 + 32 + koff);
                br[c] = *(const v8h*)(gB + (size_t)row * Kd + k0 + 32 + koff);
            }
        }
        mma_tile(&sA[0][0], &sB[0][0], wm, wn, lrow, lhi, acc);
        __syncthreads();
    }
#endif

    // ---- store: D element (m = r + 8*lhi, n = lrow) per VGPR r ----
    #pragma unroll
    for (int mt = 0; mt < 2; mt++) {
        const int mrow0 = m0 + wm * 32 + mt * 16 + lhi * 8;
        const int ncol0 = n0 + wn * 64 + lrow;
        #pragma unroll
        for (int nt = 0; nt < 4; nt++) {
            int ncol = ncol0 + nt * 16;
            #pragma unroll
            for (int r = 0; r < 8; r++) {
                int m = mrow0 + r;
                float v = acc[mt][nt][r] * outScale;
                size_t idx = TRANSOUT ? ((size_t)ncol * M + m) : ((size_t)m * Nc + ncol);
                if (OUT16) {
                    ((half_t*)Out + (size_t)bz * oBatch)[idx] = (half_t)v;
                } else if (EPILOGUE) {
                    ((float*)Out + (size_t)bz * oBatch)[idx] =
                        v + bias[m] + resid[(size_t)bz * rBatch + idx];
                } else {
                    ((float*)Out)[idx] = v;
                }
            }
        }
    }
}

// ---------------------------------------------------------------------------
// Row softmax: P[i][j] = softmax_j(S[i][:]) f16
// ---------------------------------------------------------------------------
__global__ __launch_bounds__(256) void softmax_row(const float* __restrict__ S,
                                                   half_t* __restrict__ P, int n) {
    int i = blockIdx.x;
    const float* row = S + (size_t)i * n;
    float m = -1e30f;
    for (int j = threadIdx.x; j < n; j += blockDim.x) m = fmaxf(m, row[j]);
    m = block_reduce(m, true);
    float l = 0.f;
    for (int j = threadIdx.x; j < n; j += blockDim.x) l += __expf(row[j] - m);
    l = block_reduce(l, false);
    float inv = 1.0f / l;
    half_t* prow = P + (size_t)i * n;
    for (int j = threadIdx.x; j < n; j += blockDim.x)
        prow[j] = (half_t)(__expf(row[j] - m) * inv);
}

// ---------------------------------------------------------------------------
// host orchestration
// ---------------------------------------------------------------------------
extern "C" void kernel_launch(void* const* d_in, const int* in_sizes, int n_in,
                              void* d_out, int out_size, void* d_ws, size_t ws_size,
                              hipStream_t stream) {
    const float* x     = (const float*)d_in[0];
    const float* gamma = (const float*)d_in[1];
    const float* beta  = (const float*)d_in[2];
    const float* wq    = (const float*)d_in[3];
    const float* wk    = (const float*)d_in[4];
    const float* wv    = (const float*)d_in[5];
    const float* wp    = (const float*)d_in[6];
    const float* bp    = (const float*)d_in[7];
    float* out = (float*)d_out;

    char* ws = (char*)d_ws;
    size_t off = 0;
    auto take = [&](size_t bytes) -> char* {
        char* p = ws + off;
        off += (bytes + 255) & ~(size_t)255;
        return p;
    };
    half_t* Wq16 = (half_t*)take((size_t)Cc * Cc * 2);
    half_t* Wk16 = (half_t*)take((size_t)Cc * Cc * 2);
    half_t* Wv16 = (half_t*)take((size_t)Cc * Cc * 2);
    half_t* Wp16 = (half_t*)take((size_t)Cc * Cc * 2);
    float*  meanb = (float*)take((size_t)Bn * Gg * 4);
    float*  rstdb = (float*)take((size_t)Bn * Gg * 4);
    half_t* Zt = (half_t*)take((size_t)Bn * CNs * 2);   // [b][n][c]
    half_t* Qt = (half_t*)take((size_t)Bn * CNs * 2);   // [b][i][c]
    half_t* Kt = (half_t*)take((size_t)Bn * CNs * 2);   // [b][j][c]
    half_t* V  = (half_t*)take((size_t)Bn * CNs * 2);   // [b][c][j]
    half_t* Ot = (half_t*)take((size_t)Bn * CNs * 2);   // [b][i][c]
    float*  S  = (float*) take((size_t)Nn * Nn * 4);    // per-batch scratch
    half_t* P  = (half_t*)take((size_t)Nn * Nn * 2);    // per-batch scratch

    const int we = Cc * Cc;
    f32_to_f16<<<(we + 255) / 256, 256, 0, stream>>>(wq, Wq16, we);
    f32_to_f16<<<(we + 255) / 256, 256, 0, stream>>>(wk, Wk16, we);
    f32_to_f16<<<(we + 255) / 256, 256, 0, stream>>>(wv, Wv16, we);
    f32_to_f16<<<(we + 255) / 256, 256, 0, stream>>>(wp, Wp16, we);

    gn_stats<<<Bn * Gg, 256, 0, stream>>>(x, meanb, rstdb);
    size_t total = (size_t)Bn * CNs;
    gn_apply<<<(unsigned)((total + 255) / 256), 256, 0, stream>>>(
        x, meanb, rstdb, gamma, beta, Zt);

    // QKV: A=W16[o][c], B=Zt[n][c]; Q/K stored transposed ([token][chan])
    dim3 gqkv(Nn / 128, Cc / 128, Bn);
    gemm_wmma<true, false, true><<<gqkv, 256, 0, stream>>>(
        Wq16, Zt, Qt, Cc, Nn, Cc, 0, (long long)CNs, (long long)CNs,
        0.04419417382f, nullptr, nullptr, 0);
    gemm_wmma<true, false, true><<<gqkv, 256, 0, stream>>>(
        Wk16, Zt, Kt, Cc, Nn, Cc, 0, (long long)CNs, (long long)CNs,
        1.0f, nullptr, nullptr, 0);
    gemm_wmma<true, false, false><<<gqkv, 256, 0, stream>>>(
        Wv16, Zt, V, Cc, Nn, Cc, 0, (long long)CNs, (long long)CNs,
        1.0f, nullptr, nullptr, 0);

    for (int b = 0; b < Bn; b++) {
        // S[i][j] = sum_c Qt[i][c] * Kt[j][c]
        dim3 gs(Nn / 128, Nn / 128, 1);
        gemm_wmma<false, false, false><<<gs, 256, 0, stream>>>(
            Qt + (size_t)b * CNs, Kt + (size_t)b * CNs, S, Nn, Nn, Cc,
            0, 0, 0, 1.0f, nullptr, nullptr, 0);

        softmax_row<<<Nn, 256, 0, stream>>>(S, P, Nn);

        // O[c][i] = sum_j V[c][j] * P[i][j]; stored transposed -> Ot[i][c]
        dim3 go(Nn / 128, Cc / 128, 1);
        gemm_wmma<true, false, true><<<go, 256, 0, stream>>>(
            V + (size_t)b * CNs, P, Ot + (size_t)b * CNs, Cc, Nn, Nn,
            0, 0, 0, 1.0f, nullptr, nullptr, 0);
    }

    // out[c][n] = x + sum_c' Wp[c][c'] * Ot[n][c'] + bp[c]
    dim3 gf(Nn / 128, Cc / 128, Bn);
    gemm_wmma<false, true, false><<<gf, 256, 0, stream>>>(
        Wp16, Ot, out, Cc, Nn, Cc, 0, (long long)CNs, (long long)CNs,
        1.0f, bp, x, (long long)CNs);
}
